// MultiHeadAttention_22557168239110
// MI455X (gfx1250) — matrix operand order repaired and checked by reading, not compiled, not run on hardware
//
#include <hip/hip_runtime.h>
#include <hip/hip_bf16.h>
#include <cstdint>

typedef __attribute__((ext_vector_type(16))) _Float16 v16h;
typedef __attribute__((ext_vector_type(8)))  float    v8f;
typedef __attribute__((ext_vector_type(4)))  unsigned int v4u;
typedef __attribute__((ext_vector_type(8)))  int      v8i_;
typedef __attribute__((ext_vector_type(4)))  int      v4i_;

#define DIM   1024
#define HEADS 16
#define HD    64
#define SEQ   2048
#define BATCH 2

// ---------------------------------------------------------------------------
// WMMA fragment gathers (wave32, V_WMMA_F32_16X16X32_F16 A/B in contiguous k = 16*(lane/16) + s order)
// ---------------------------------------------------------------------------

// A-matrix 16x32 f16, row-major source with stride ld (elements).
__device__ __forceinline__ v16h load_a_frag(const _Float16* t, int ld) {
  const int lane = threadIdx.x & 31;
  const _Float16* row = t + (lane & 15) * ld;
  const int kb = (lane >> 4) << 4;  // 0 or 16
  v16h a;
#pragma unroll
  for (int j = 0; j < 4; ++j) {
    a[2 * j]         = row[kb + 2 * j];
    a[2 * j + 1]     = row[kb + 2 * j + 1];
    a[8 + 2 * j]     = row[kb + 8 + 2 * j];
    a[8 + 2 * j + 1] = row[kb + 8 + 2 * j + 1];
  }
  return a;
}

// B-matrix 32x16 f16 from k-major tile t[k*ld + n].
__device__ __forceinline__ v16h load_b_frag_kmajor(const _Float16* t, int ld) {
  const int lane = threadIdx.x & 31;
  const int n  = lane & 15;
  const int kb = (lane >> 4) << 4;  // 0 or 16
  v16h b;
#pragma unroll
  for (int j = 0; j < 8; ++j) {
    b[2 * j]     = t[(kb + 2 * j) * ld + n];
    b[2 * j + 1] = t[(kb + 2 * j + 1) * ld + n];
  }
  return b;
}

// B-matrix 32x16 where the source tile is n-major (t[n*ld + k]) -> B = tile^T.
__device__ __forceinline__ v16h load_b_frag_nmajor(const _Float16* t, int ld) {
  const int lane = threadIdx.x & 31;
  const _Float16* row = t + (lane & 15) * ld;
  const int kb = (lane >> 4) << 4;  // 0 or 16
  v16h b;
#pragma unroll
  for (int j = 0; j < 8; ++j) {
    b[2 * j]     = row[kb + 2 * j];
    b[2 * j + 1] = row[kb + 2 * j + 1];
  }
  return b;
}

__device__ __forceinline__ v8f wmma_f16(v16h a, v16h b, v8f c) {
  return __builtin_amdgcn_wmma_f32_16x16x32_f16(false, a, false, b,
                                                (short)0, c, false, false);
}

// ---------------------------------------------------------------------------
// TDM: DMA a 64x64 f16 tile (row stride 64 elems in memory) into LDS with a
// 66-half row stride (pad 1 DWORD after every 32 DWORDs). D# per ISA ch.8.
// One call per wave moves the whole tile; tracked by TENSORcnt.
// This toolchain exposes the 6-arg builtin:
//   (uint32x4 g0, int32x8 g1, int32x4, int32x4, int32x8, i32 cpol)
// ---------------------------------------------------------------------------
__device__ __forceinline__ void tdm_load_tile_64x64(const _Float16* gsrc,
                                                    _Float16* lds_dst) {
  const unsigned long long ga = (unsigned long long)(uintptr_t)gsrc;
  const unsigned int la = (unsigned int)(uintptr_t)lds_dst;  // low 32b = LDS offset
  v4u g0;
  g0.x = 0x1u;                                   // count=1, user descriptor
  g0.y = la;                                     // lds_addr (bytes)
  g0.z = (unsigned int)(ga & 0xffffffffu);       // global_addr[31:0]
  g0.w = (unsigned int)((ga >> 32) & 0x01ffffffu) | 0x80000000u;  // [56:32] | type=2
  v8i_ g1;
  g1[0] = (int)(0x00010000u      // data_size = 1 -> 2 bytes
              | 0x00100000u      // pad_enable
              | (4u << 22));     // pad_interval = 4 -> every 32 DWORDs, pad_amount=0 -> 1 DWORD
  g1[1] = (int)(64u << 16);      // tensor_dim0[15:0] = 64 (bits 63:48 of group)
  g1[2] = (int)(2048u << 16);    // tensor_dim1[15:0] = SEQ
  g1[3] = (int)(64u << 16);      // tile_dim0 = 64
  g1[4] = 64;                    // tile_dim1 = 64 (tile_dim2 = 0)
  g1[5] = 64;                    // tensor_dim0_stride = 64
  g1[6] = 0;
  g1[7] = 0;
  const v4i_ zz4 = {0, 0, 0, 0};              // groups 2/3 unused (2-D tensor)
  const v8i_ zz8 = {0, 0, 0, 0, 0, 0, 0, 0};  // extra group (unused)
  __builtin_amdgcn_tensor_load_to_lds(g0, g1, zz4, zz4, zz8, 0);
}

// ---------------------------------------------------------------------------
// GEMM: Y[4096,1024] = X @ W^T + bias. 128x128x32 tile, 256 thr = 8 waves (2x4),
// wave tile 64x32. Double-buffered LDS with register staging: global loads for
// tile t+1 issue before the WMMAs on tile t; one barrier per k-iteration.
// ---------------------------------------------------------------------------
template <bool IN_F32, bool OUT_HEADSPLIT>
__global__ __launch_bounds__(256)
void gemm128(const void* __restrict__ Xv, const float* __restrict__ W,
             const float* __restrict__ bias, void* __restrict__ Yv) {
  constexpr int BK = 32;
  constexpr int LDA = 36;   // 72B rows: 8B aligned, odd-word stride
  constexpr int LDB = 136;
  __shared__ __align__(16) _Float16 As[2][128 * LDA];
  __shared__ __align__(16) _Float16 Bs[2][BK * LDB];

  const int tid  = threadIdx.x;
  const int wave = tid >> 5;
  const int lane = tid & 31;
  const int m0 = blockIdx.y * 128;
  const int n0 = blockIdx.x * 128;
  const int wm = (wave >> 2) * 64;
  const int wn = (wave & 3) * 32;

  const float*    Xf = (const float*)Xv;
  const _Float16* Xh = (const _Float16*)Xv;

  const int sr = tid >> 3;        // staging row 0..31 (+i*32)
  const int sc = (tid & 7) * 4;   // staging col 0..28

  float4 areg[4]; uint2 aregh[4]; float4 breg[4];

  auto stage_load = [&](int k0) {
#pragma unroll
    for (int i = 0; i < 4; ++i) {
      const int rr = sr + i * 32;
      if (IN_F32) areg[i]  = *(const float4*)(Xf + (size_t)(m0 + rr) * DIM + k0 + sc);
      else        aregh[i] = *(const uint2*)(Xh + (size_t)(m0 + rr) * DIM + k0 + sc);
      breg[i] = *(const float4*)(W + (size_t)(n0 + rr) * DIM + k0 + sc);
    }
  };
  auto stage_store = [&](int buf) {
#pragma unroll
    for (int i = 0; i < 4; ++i) {
      const int rr = sr + i * 32;
      _Float16* da = &As[buf][rr * LDA + sc];
      if (IN_F32) {
        da[0] = (_Float16)areg[i].x; da[1] = (_Float16)areg[i].y;
        da[2] = (_Float16)areg[i].z; da[3] = (_Float16)areg[i].w;
      } else {
        ((uint32_t*)da)[0] = aregh[i].x;  // 4B aligned (72B rows, sc%4==0)
        ((uint32_t*)da)[1] = aregh[i].y;
      }
      Bs[buf][(sc + 0) * LDB + rr] = (_Float16)breg[i].x;
      Bs[buf][(sc + 1) * LDB + rr] = (_Float16)breg[i].y;
      Bs[buf][(sc + 2) * LDB + rr] = (_Float16)breg[i].z;
      Bs[buf][(sc + 3) * LDB + rr] = (_Float16)breg[i].w;
    }
  };

  v8f acc[4][2] = {};

  stage_load(0);
  stage_store(0);
  __syncthreads();

  int cur = 0;
  for (int k0 = 0; k0 < DIM; k0 += BK) {
    const bool has_next = (k0 + BK < DIM);
    if (has_next) {
      stage_load(k0 + BK);  // issue early; latency hidden under WMMAs below
      if (k0 + 2 * BK < DIM) {
        const size_t esz = IN_F32 ? 4 : 2;
        __builtin_prefetch((const char*)Xv +
                           ((size_t)(m0 + sr) * DIM + k0 + 2 * BK) * esz, 0, 1);
        __builtin_prefetch(W + (size_t)(n0 + sr) * DIM + k0 + 2 * BK, 0, 1);
      }
    }

    v16h af[4], bf[2];
#pragma unroll
    for (int mi = 0; mi < 4; ++mi)
      af[mi] = load_a_frag(&As[cur][(wm + mi * 16) * LDA], LDA);
#pragma unroll
    for (int ni = 0; ni < 2; ++ni)
      bf[ni] = load_b_frag_kmajor(&Bs[cur][wn + ni * 16], LDB);
#pragma unroll
    for (int mi = 0; mi < 4; ++mi)
#pragma unroll
      for (int ni = 0; ni < 2; ++ni)
        acc[mi][ni] = wmma_f16(af[mi], bf[ni], acc[mi][ni]);

    if (has_next) stage_store(cur ^ 1);
    __syncthreads();
    cur ^= 1;
  }

  // ---- epilogue: bias + store (C/D layout: VGPR r -> row r + 8*(lane>=16)) ----
  const int half = lane >> 4;
  const int nl   = lane & 15;
#pragma unroll
  for (int mi = 0; mi < 4; ++mi) {
#pragma unroll
    for (int ni = 0; ni < 2; ++ni) {
      const int ncol = n0 + wn + ni * 16 + nl;
      const float bv = bias[ncol];
#pragma unroll
      for (int r = 0; r < 8; ++r) {
        const int mrow = m0 + wm + mi * 16 + r + half * 8;
        const float val = acc[mi][ni][r] + bv;
        if (OUT_HEADSPLIT) {
          const int b = mrow >> 11, s = mrow & (SEQ - 1);
          const int h = ncol >> 6,  d = ncol & (HD - 1);
          ((_Float16*)Yv)[(((size_t)b * HEADS + h) * SEQ + s) * HD + d] =
              (_Float16)val;
        } else {
          ((float*)Yv)[(size_t)mrow * DIM + ncol] = val;
        }
      }
    }
  }
}

// ---------------------------------------------------------------------------
// Flash attention: one block per (64-query tile, b*h). 4 waves, 16 q-rows each.
// K/V tiles are DMA'd by the Tensor Data Mover into double-buffered LDS
// (wave 0 issues, completion via s_wait_tensorcnt), overlapping the next
// tile's DMA with this tile's WMMAs + online softmax.
// ---------------------------------------------------------------------------
__global__ __launch_bounds__(128)
void flash_attn(const _Float16* __restrict__ Qh, const _Float16* __restrict__ Kh,
                const _Float16* __restrict__ Vh, _Float16* __restrict__ Oh,
                const int* __restrict__ causal_p) {
  constexpr int LD = 66;  // 132B rows: odd-word stride, conflict-free
  __shared__ __align__(16) _Float16 Qt[64 * LD];
  __shared__ __align__(16) _Float16 Kt[2][64 * LD];
  __shared__ __align__(16) _Float16 Vt[2][64 * LD];
  __shared__ __align__(16) _Float16 Pt[64 * LD];  // 16 rows per wave, wave-private

  const int tid  = threadIdx.x;
  const int wave = tid >> 5;
  const int lane = tid & 31;
  const int half = lane >> 4;
  const int nl   = lane & 15;
  const int q0 = blockIdx.x * 64;
  const int bh = blockIdx.y;
  const int causal = *causal_p;

  const _Float16* Qg = Qh + (size_t)bh * SEQ * HD;
  const _Float16* Kg = Kh + (size_t)bh * SEQ * HD;
  const _Float16* Vg = Vh + (size_t)bh * SEQ * HD;

  const int kbmax = causal ? (int)blockIdx.x : (SEQ / 64 - 1);

  // TDM prologue: start DMA of the first K/V tiles.
  if (wave == 0) {
    tdm_load_tile_64x64(Kg, &Kt[0][0]);
    tdm_load_tile_64x64(Vg, &Vt[0][0]);
  }

  // ---- stage Q tile (64x64 f16) with vector loads, keep as register A-frags ----
  {
    const int r  = tid >> 1;
    const int cb = (tid & 1) * 32;
#pragma unroll
    for (int i = 0; i < 4; ++i) {
      const int c = cb + i * 8;
      const uint4 v = *(const uint4*)(Qg + (size_t)(q0 + r) * HD + c);
      uint32_t* d = (uint32_t*)&Qt[r * LD + c];
      d[0] = v.x; d[1] = v.y; d[2] = v.z; d[3] = v.w;
    }
  }
  __syncthreads();
  v16h qf[2];
  qf[0] = load_a_frag(&Qt[(wave * 16) * LD + 0], LD);
  qf[1] = load_a_frag(&Qt[(wave * 16) * LD + 32], LD);

  v8f oacc[4] = {};
  float mrow[8], lrow[8];
#pragma unroll
  for (int r = 0; r < 8; ++r) { mrow[r] = -3.0e38f; lrow[r] = 0.0f; }

  for (int kb = 0; kb <= kbmax; ++kb) {
    const int cur = kb & 1;
    if (wave == 0) {
      if (kb < kbmax) {
        // issue next tile's DMA, then wait for the current tile's 2 loads
        // (TDM ops complete in order: outstanding <= 2 => block kb is done)
        tdm_load_tile_64x64(Kg + (size_t)(kb + 1) * 64 * HD, &Kt[cur ^ 1][0]);
        tdm_load_tile_64x64(Vg + (size_t)(kb + 1) * 64 * HD, &Vt[cur ^ 1][0]);
        __builtin_amdgcn_s_wait_tensorcnt(2);
      } else {
        __builtin_amdgcn_s_wait_tensorcnt(0);
      }
    }
    __syncthreads();  // tile kb visible to all waves

    // ---- S = Q * K^T (4 frags of 16x16 over the 64 keys) ----
    v8f sf[4];
#pragma unroll
    for (int nf = 0; nf < 4; ++nf) {
      v8f s = {};
      s = wmma_f16(qf[0], load_b_frag_nmajor(&Kt[cur][(nf * 16) * LD + 0], LD), s);
      s = wmma_f16(qf[1], load_b_frag_nmajor(&Kt[cur][(nf * 16) * LD + 32], LD), s);
      sf[nf] = s;
    }

    // ---- scale, causal mask, row max ----
    const float scale = 0.125f;  // 1/sqrt(64)
    float rmax[8];
#pragma unroll
    for (int r = 0; r < 8; ++r) rmax[r] = -3.0e38f;
#pragma unroll
    for (int nf = 0; nf < 4; ++nf) {
#pragma unroll
      for (int r = 0; r < 8; ++r) {
        float v = sf[nf][r] * scale;
        if (causal) {
          const int qrow = q0 + wave * 16 + r + half * 8;
          const int kcol = kb * 64 + nf * 16 + nl;
          if (kcol > qrow) v = -3.0e38f;
        }
        sf[nf][r] = v;
        rmax[r] = fmaxf(rmax[r], v);
      }
    }
#pragma unroll
    for (int r = 0; r < 8; ++r)
#pragma unroll
      for (int off = 8; off >= 1; off >>= 1)
        rmax[r] = fmaxf(rmax[r], __shfl_xor(rmax[r], off, 16));

    // ---- online softmax update ----
    float alpha[8], mnew[8], rsum[8];
#pragma unroll
    for (int r = 0; r < 8; ++r) {
      mnew[r] = fmaxf(mrow[r], rmax[r]);
      alpha[r] = __expf(mrow[r] - mnew[r]);
      mrow[r] = mnew[r];
      rsum[r] = 0.0f;
    }
#pragma unroll
    for (int nf = 0; nf < 4; ++nf) {
#pragma unroll
      for (int r = 0; r < 8; ++r) {
        const float p = __expf(sf[nf][r] - mnew[r]);
        rsum[r] += p;
        Pt[(wave * 16 + r + half * 8) * LD + nf * 16 + nl] = (_Float16)p;
      }
    }
#pragma unroll
    for (int r = 0; r < 8; ++r) {
#pragma unroll
      for (int off = 8; off >= 1; off >>= 1)
        rsum[r] += __shfl_xor(rsum[r], off, 16);
      lrow[r] = lrow[r] * alpha[r] + rsum[r];
    }
#pragma unroll
    for (int j = 0; j < 4; ++j)
#pragma unroll
      for (int r = 0; r < 8; ++r) oacc[j][r] *= alpha[r];

    // ---- O += P * V ---- (Pt is wave-private; DS ops are in-order per wave)
    const v16h pf0 = load_a_frag(&Pt[(wave * 16) * LD + 0], LD);
    const v16h pf1 = load_a_frag(&Pt[(wave * 16) * LD + 32], LD);
#pragma unroll
    for (int j = 0; j < 4; ++j) {
      oacc[j] = wmma_f16(pf0, load_b_frag_kmajor(&Vt[cur][0 * LD + j * 16], LD), oacc[j]);
      oacc[j] = wmma_f16(pf1, load_b_frag_kmajor(&Vt[cur][32 * LD + j * 16], LD), oacc[j]);
    }
    __syncthreads();  // all waves done with tile kb before its buffer is re-DMA'd
  }

  // ---- normalize and store O as f16 in [b, s, h*64+d] for the final GEMM ----
  const int b = bh >> 4, h = bh & 15;
#pragma unroll
  for (int r = 0; r < 8; ++r) {
    const float inv = 1.0f / lrow[r];
    const int s = q0 + wave * 16 + r + half * 8;
#pragma unroll
    for (int j = 0; j < 4; ++j) {
      const int col = h * HD + j * 16 + nl;
      Oh[((size_t)b * SEQ + s) * DIM + col] = (_Float16)(oacc[j][r] * inv);
    }
  }
}

// ---------------------------------------------------------------------------
// Host launcher. Workspace: Qh, Kh, Vh f16 [B,H,S,HD] + attn-out f16 [B,S,DIM]
// = 4 * 8 MB = 32 MB.
// ---------------------------------------------------------------------------
extern "C" void kernel_launch(void* const* d_in, const int* in_sizes, int n_in,
                              void* d_out, int out_size, void* d_ws, size_t ws_size,
                              hipStream_t stream) {
  const float* Q_in = (const float*)d_in[0];
  const float* K_in = (const float*)d_in[1];
  const float* V_in = (const float*)d_in[2];
  const float* Wq = (const float*)d_in[3];
  const float* bq = (const float*)d_in[4];
  const float* Wk = (const float*)d_in[5];
  const float* bk = (const float*)d_in[6];
  const float* Wv = (const float*)d_in[7];
  const float* bv = (const float*)d_in[8];
  const float* Wo = (const float*)d_in[9];
  const float* bo = (const float*)d_in[10];
  const int* is_causal = (const int*)d_in[11];
  (void)in_sizes; (void)n_in; (void)out_size; (void)ws_size;

  const size_t elems = (size_t)BATCH * SEQ * DIM;  // 4M f16 each
  _Float16* Qh = (_Float16*)d_ws;
  _Float16* Kh = Qh + elems;
  _Float16* Vh = Kh + elems;
  _Float16* Oh = Vh + elems;

  dim3 gg(DIM / 128, (BATCH * SEQ) / 128);  // (8, 32)
  gemm128<true, true><<<gg, 256, 0, stream>>>(Q_in, Wq, bq, Qh);
  gemm128<true, true><<<gg, 256, 0, stream>>>(K_in, Wk, bk, Kh);
  gemm128<true, true><<<gg, 256, 0, stream>>>(V_in, Wv, bv, Vh);

  dim3 ga(SEQ / 64, BATCH * HEADS);  // (32, 32)
  flash_attn<<<ga, 128, 0, stream>>>(Qh, Kh, Vh, Oh, is_causal);

  gemm128<false, false><<<gg, 256, 0, stream>>>(Oh, Wo, bo, d_out);
}
